// GNNDenoiser_35725537968359
// MI455X (gfx1250) — matrix operand order repaired and checked by reading, not compiled, not run on hardware
//
#include <hip/hip_runtime.h>
#include <hip/hip_fp16.h>
#include <hip/hip_bf16.h>

typedef __attribute__((ext_vector_type(16))) _Float16 v16h;
typedef __attribute__((ext_vector_type(8)))  float    v8f;

#define NNODE   50000
#define NEDGE   600000
#define NHEAD   25000
#define DIM     128

// ---------------------------------------------------------------------------
// WMMA helpers (gfx1250, wave32). Layouts per CDNA5 ISA 7.12.2.
// ---------------------------------------------------------------------------
__device__ inline v8f wmma32(v16h a, v16h b, v8f c) {
  // v_wmma_f32_16x16x32_f16  D = A(16x32) * B(32x16) + C(16x16 f32)
  return __builtin_amdgcn_wmma_f32_16x16x32_f16(
      /*neg_a=*/false, a, /*neg_b=*/false, b,
      /*c_mod=*/(short)0, c, /*reuse_a=*/false, /*reuse_b=*/false);
}

__device__ inline v8f zero8() {
  v8f z;
#pragma unroll
  for (int q = 0; q < 8; ++q) z[q] = 0.0f;
  return z;
}

// A fragment: 16x32 f16 tile, row-major in LDS with leading dim lda.
// Lanes 0-15: M=lane, VGPR0-3 hold K=0..7, VGPR4-7 hold K=16..23.
// Lanes 16-31: same M, K offset +8 (first group) / +8 (second group).
__device__ inline v16h load_a_frag(const _Float16* p, int lda) {
  const int lane = threadIdx.x & 31;
  const int m  = lane & 15;
  const int hh = lane >> 4;
  v16h a;
#pragma unroll
  for (int v = 0; v < 8; ++v) {
    const int k0 = ((v >> 2) << 4) + (hh << 3) + ((v & 3) << 1);
    a[2 * v]     = p[m * lda + k0];
    a[2 * v + 1] = p[m * lda + k0 + 1];
  }
  return a;
}

// B fragment: 32x16 f16 tile staged transposed (n-major) in LDS: WT[n][k],
// leading dim ldk. Lanes 0-15: VGPR0-3 hold K=0..7, VGPR4-7 hold K=16..23;
// lanes 16-31: K offset +8 (same order as the A fragment).
__device__ inline v16h load_b_frag(const _Float16* p, int ldk) {
  const int lane = threadIdx.x & 31;
  const int n  = lane & 15;
  const int hh = lane >> 4;
  v16h b;
#pragma unroll
  for (int v = 0; v < 8; ++v) {
    const int k0 = ((v >> 2) << 4) + (hh << 3) + ((v & 3) << 1);
    b[2 * v]     = p[n * ldk + k0];
    b[2 * v + 1] = p[n * ldk + k0 + 1];
  }
  return b;
}

// ---------------------------------------------------------------------------
// Kernel 1: fused node init  h0 = x @ projW + projB + timeMLP(t)
//   timeMLP(t) = silu([cos|sin](t) @ W1 + b1) @ W2 + b2
// Block = 128 threads (4 waves), 64 nodes per block, wave owns 16 rows.
// ---------------------------------------------------------------------------
template <int DX>
__global__ void __launch_bounds__(128)
k_node_init(const float* __restrict__ x, const int* __restrict__ t,
            const float* __restrict__ te_W1, const float* __restrict__ te_b1,
            const float* __restrict__ te_W2, const float* __restrict__ te_b2,
            const float* __restrict__ proj_W, const float* __restrict__ proj_b,
            float* __restrict__ h0, int M) {
  extern __shared__ _Float16 sm[];
  _Float16* W1T = sm;                  // 128*128  (n-major)
  _Float16* W2T = W1T + 128 * 128;     // 128*128
  _Float16* PWT = W2T + 128 * 128;     // 128*DX
  _Float16* EMB = PWT + 128 * DX;      // 64*128
  _Float16* H1  = EMB + 64 * 128;      // 64*128
  _Float16* XT  = H1 + 64 * 128;       // 64*DX

  const int tid = threadIdx.x;
  const int n0  = blockIdx.x * 64;

  for (int i = tid; i < 128 * 128; i += 128) {
    const int n = i >> 7, k = i & 127;
    W1T[n * 128 + k] = (_Float16)te_W1[k * 128 + n];
    W2T[n * 128 + k] = (_Float16)te_W2[k * 128 + n];
  }
  for (int i = tid; i < 128 * DX; i += 128) {
    const int n = i / DX, k = i - n * DX;
    PWT[i] = (_Float16)proj_W[k * 128 + n];
  }
  const float lnp = 9.210340371976184f;  // ln(10000)
  for (int i = tid; i < 64 * 128; i += 128) {
    const int r = i >> 7, j = i & 127;
    int node = n0 + r; if (node >= M) node = M - 1;
    const float tt = (float)t[node];
    const int jj   = (j < 64) ? j : (j - 64);
    const float fr = __expf(-lnp * (float)jj / 64.0f);
    const float ar = tt * fr;
    EMB[i] = (_Float16)((j < 64) ? __cosf(ar) : __sinf(ar));
  }
  for (int i = tid; i < 64 * DX; i += 128) {
    const int r = i / DX, k = i - r * DX;
    int node = n0 + r; if (node >= M) node = M - 1;
    XT[i] = (_Float16)x[node * DX + k];
  }
  __syncthreads();

  const int wave = tid >> 5, lane = tid & 31;
  const int r0 = wave * 16, cn = lane & 15, ch = lane >> 4;

  // GEMM1: H1 = silu(EMB @ W1 + b1)
#pragma unroll
  for (int nt = 0; nt < 8; ++nt) {
    v8f acc = zero8();
#pragma unroll
    for (int kt = 0; kt < 4; ++kt) {
      v16h a = load_a_frag(&EMB[r0 * 128 + kt * 32], 128);
      v16h b = load_b_frag(&W1T[(nt * 16) * 128 + kt * 32], 128);
      acc = wmma32(a, b, acc);
    }
    const int n = nt * 16 + cn;
    const float bias = te_b1[n];
#pragma unroll
    for (int v = 0; v < 8; ++v) {
      const int m = ch * 8 + v;
      const float z = acc[v] + bias;
      H1[(r0 + m) * 128 + n] = (_Float16)(z / (1.0f + __expf(-z)));
    }
  }
  __syncthreads();

  // GEMM2 (+ bias b2) and projection GEMM fused into one accumulator.
#pragma unroll
  for (int nt = 0; nt < 8; ++nt) {
    v8f acc = zero8();
#pragma unroll
    for (int kt = 0; kt < 4; ++kt) {
      v16h a = load_a_frag(&H1[r0 * 128 + kt * 32], 128);
      v16h b = load_b_frag(&W2T[(nt * 16) * 128 + kt * 32], 128);
      acc = wmma32(a, b, acc);
    }
#pragma unroll
    for (int kt = 0; kt < DX / 32; ++kt) {
      v16h a = load_a_frag(&XT[r0 * DX + kt * 32], DX);
      v16h b = load_b_frag(&PWT[(nt * 16) * DX + kt * 32], DX);
      acc = wmma32(a, b, acc);
    }
    const int n = nt * 16 + cn;
    const float bias = te_b2[n] + proj_b[n];
#pragma unroll
    for (int v = 0; v < 8; ++v) {
      const int node = n0 + r0 + ch * 8 + v;
      if (node < M) h0[node * 128 + n] = acc[v] + bias;
    }
  }
}

// ---------------------------------------------------------------------------
// Kernel 2: zero scratch
// ---------------------------------------------------------------------------
__global__ void k_zero(float* __restrict__ p, long long n) {
  long long i = (long long)blockIdx.x * blockDim.x + threadIdx.x;
  const long long st = (long long)gridDim.x * blockDim.x;
  for (; i < n; i += st) p[i] = 0.0f;
}

// ---------------------------------------------------------------------------
// Kernel 3: edge scatter-add (segment sum + degree count).
// One thread per (edge, 4 channels): float4 gather + 4 f32 global atomics.
// ---------------------------------------------------------------------------
__global__ void __launch_bounds__(256)
k_scatter(const float* __restrict__ h, const int* __restrict__ src,
          const int* __restrict__ dst, float* __restrict__ meanb,
          float* __restrict__ cnt, int E) {
  const long long gid = (long long)blockIdx.x * 256 + threadIdx.x;
  const int e = (int)(gid >> 5);
  if (e >= E) return;
  const int lane = (int)(gid & 31);
  const int s = src[e], d = dst[e];
  const int c = lane * 4;
  const float4 v = *(const float4*)(h + (long long)s * 128 + c);
  float* mp = meanb + (long long)d * 128 + c;
  unsafeAtomicAdd(mp + 0, v.x);
  unsafeAtomicAdd(mp + 1, v.y);
  unsafeAtomicAdd(mp + 2, v.z);
  unsafeAtomicAdd(mp + 3, v.w);
  if (lane == 0) unsafeAtomicAdd(cnt + d, 1.0f);
}

// ---------------------------------------------------------------------------
// Kernel 4: SAGE transform  out = relu((sum/deg) @ Wn + h_dst @ Wr + b)
// ---------------------------------------------------------------------------
__global__ void __launch_bounds__(128)
k_sage(const float* __restrict__ meanb, const float* __restrict__ cnt,
       const float* __restrict__ hdst, const float* __restrict__ Wn,
       const float* __restrict__ Wr, const float* __restrict__ bias,
       float* __restrict__ out, int M) {
  extern __shared__ _Float16 sm[];
  _Float16* WnT = sm;                 // 128*128
  _Float16* WrT = WnT + 128 * 128;    // 128*128
  _Float16* MT  = WrT + 128 * 128;    // 64*128
  _Float16* DT  = MT + 64 * 128;      // 64*128

  const int tid = threadIdx.x;
  const int n0  = blockIdx.x * 64;

  for (int i = tid; i < 128 * 128; i += 128) {
    const int n = i >> 7, k = i & 127;
    WnT[n * 128 + k] = (_Float16)Wn[k * 128 + n];
    WrT[n * 128 + k] = (_Float16)Wr[k * 128 + n];
  }
  for (int i = tid; i < 64 * 128; i += 128) {
    const int r = i >> 7, k = i & 127;
    int node = n0 + r; if (node >= M) node = M - 1;
    const float inv = 1.0f / fmaxf(cnt[node], 1.0f);
    MT[i] = (_Float16)(meanb[node * 128 + k] * inv);
    DT[i] = (_Float16)hdst[node * 128 + k];
  }
  __syncthreads();

  const int wave = tid >> 5, lane = tid & 31;
  const int r0 = wave * 16, cn = lane & 15, ch = lane >> 4;
#pragma unroll
  for (int nt = 0; nt < 8; ++nt) {
    v8f acc = zero8();
#pragma unroll
    for (int kt = 0; kt < 4; ++kt)
      acc = wmma32(load_a_frag(&MT[r0 * 128 + kt * 32], 128),
                   load_b_frag(&WnT[(nt * 16) * 128 + kt * 32], 128), acc);
#pragma unroll
    for (int kt = 0; kt < 4; ++kt)
      acc = wmma32(load_a_frag(&DT[r0 * 128 + kt * 32], 128),
                   load_b_frag(&WrT[(nt * 16) * 128 + kt * 32], 128), acc);
    const int n = nt * 16 + cn;
    const float b = bias[n];
#pragma unroll
    for (int v = 0; v < 8; ++v) {
      const int node = n0 + r0 + ch * 8 + v;
      if (node < M) out[node * 128 + n] = fmaxf(acc[v] + b, 0.0f);
    }
  }
}

// ---------------------------------------------------------------------------
// Kernel 5: generic WMMA GEMM  C = act(A[M,K] @ W[K,N] + b)
// Block = 128 threads (4 waves) -> 64 rows; NT 16-col tiles per block.
// ---------------------------------------------------------------------------
template <int NT, bool RELU>
__global__ void __launch_bounds__(128)
k_gemm(const float* __restrict__ A, const float* __restrict__ W,
       const float* __restrict__ bias, float* __restrict__ C,
       int M, int K, int N) {
  extern __shared__ _Float16 sm[];
  _Float16* AT = sm;             // 64 * 32
  _Float16* WT = AT + 64 * 32;   // (NT*16) * 32  (n-major)

  const int tid  = threadIdx.x;
  const int row0 = blockIdx.x * 64;
  const int col0 = blockIdx.y * (NT * 16);
  const int wave = tid >> 5, lane = tid & 31;
  const int r0 = wave * 16, cn = lane & 15, ch = lane >> 4;

  v8f acc[NT];
#pragma unroll
  for (int i = 0; i < NT; ++i) acc[i] = zero8();

  for (int k0 = 0; k0 < K; k0 += 32) {
    __syncthreads();
    for (int i = tid; i < 64 * 32; i += 128) {
      const int r = i >> 5, kk = i & 31;
      int row = row0 + r; if (row >= M) row = M - 1;
      AT[i] = (_Float16)A[(long long)row * K + k0 + kk];
    }
    for (int i = tid; i < NT * 16 * 32; i += 128) {
      const int c = i >> 5, kk = i & 31;
      WT[i] = (_Float16)W[(long long)(k0 + kk) * N + col0 + c];
    }
    __syncthreads();
    const v16h a = load_a_frag(&AT[r0 * 32], 32);
#pragma unroll
    for (int nt = 0; nt < NT; ++nt) {
      const v16h b = load_b_frag(&WT[(nt * 16) * 32], 32);
      acc[nt] = wmma32(a, b, acc[nt]);
    }
  }
#pragma unroll
  for (int nt = 0; nt < NT; ++nt) {
    const int n = col0 + nt * 16 + cn;
    const float bs = bias[n];
#pragma unroll
    for (int v = 0; v < 8; ++v) {
      const int row = row0 + r0 + ch * 8 + v;
      if (row < M && n < N) {
        float z = acc[nt][v] + bs;
        if (RELU) z = fmaxf(z, 0.0f);
        C[(long long)row * N + n] = z;
      }
    }
  }
}

// ---------------------------------------------------------------------------
// Host orchestration
// ---------------------------------------------------------------------------
extern "C" void kernel_launch(void* const* d_in, const int* in_sizes, int n_in,
                              void* d_out, int out_size, void* d_ws,
                              size_t ws_size, hipStream_t stream) {
  (void)in_sizes; (void)n_in; (void)out_size; (void)ws_size;

  const float* x_user   = (const float*)d_in[0];
  const float* x_item   = (const float*)d_in[1];
  const int*   t_user   = (const int*)d_in[2];
  const int*   t_item   = (const int*)d_in[3];
  const int*   e_ui_src = (const int*)d_in[4];
  const int*   e_ui_dst = (const int*)d_in[5];
  const int*   e_iu_src = (const int*)d_in[6];
  const int*   e_iu_dst = (const int*)d_in[7];
  const float* te_W1 = (const float*)d_in[8];
  const float* te_b1 = (const float*)d_in[9];
  const float* te_W2 = (const float*)d_in[10];
  const float* te_b2 = (const float*)d_in[11];
  const float* proj_Wu = (const float*)d_in[12];
  const float* proj_bu = (const float*)d_in[13];
  const float* proj_Wi = (const float*)d_in[14];
  const float* proj_bi = (const float*)d_in[15];
  const float* sage_Wnbr  = (const float*)d_in[16];  // [2][2][128][128]
  const float* sage_Wroot = (const float*)d_in[17];
  const float* sage_b     = (const float*)d_in[18];  // [2][2][128]
  const float* mlp_u_W1 = (const float*)d_in[19];
  const float* mlp_u_b1 = (const float*)d_in[20];
  const float* mlp_u_W2 = (const float*)d_in[21];
  const float* mlp_u_b2 = (const float*)d_in[22];
  const float* mlp_u_W3 = (const float*)d_in[23];
  const float* mlp_u_b3 = (const float*)d_in[24];
  const float* mlp_i_W1 = (const float*)d_in[25];
  const float* mlp_i_b1 = (const float*)d_in[26];
  const float* mlp_i_W2 = (const float*)d_in[27];
  const float* mlp_i_b2 = (const float*)d_in[28];
  const float* mlp_i_W3 = (const float*)d_in[29];
  const float* mlp_i_b3 = (const float*)d_in[30];

  float* out = (float*)d_out;

  // Workspace layout (floats).
  const long long NF = (long long)NNODE * DIM;  // 6.4M
  float* ws   = (float*)d_ws;
  float* hu0  = ws;            // node features ping
  float* hu1  = hu0 + NF;      // node features pong
  float* hi0  = hu1 + NF;
  float* hi1  = hi0 + NF;
  float* mb   = hi1 + NF;      // segment-sum buffer (reused as MLP hid1)
  float* cntb = mb + NF;       // degree counts (50000)
  float* hid1 = mb;            // 25000*256 == 50000*128, free after convs
  float* hid2 = hu1;           // free after layer loop (final h in hu0/hi0)

  const int nodeBlocks = (NNODE + 63) / 64;
  const int headBlocks = (NHEAD + 63) / 64;
  const int scatBlocks = (NEDGE * 32 + 255) / 256;

  const size_t shm_init_u = sizeof(_Float16) *
      (128 * 128 * 2 + 128 * 64 + 64 * 128 * 2 + 64 * 64);
  const size_t shm_init_i = sizeof(_Float16) *
      (128 * 128 * 2 + 128 * 32 + 64 * 128 * 2 + 64 * 32);
  const size_t shm_sage  = sizeof(_Float16) * (128 * 128 * 2 + 64 * 128 * 2);
  const size_t shm_gemm4 = sizeof(_Float16) * (64 * 32 + 4 * 16 * 32);
  const size_t shm_gemm2 = sizeof(_Float16) * (64 * 32 + 2 * 16 * 32);

  // Stage 1: fused embedding + time-MLP + projection.
  k_node_init<64><<<nodeBlocks, 128, shm_init_u, stream>>>(
      x_user, t_user, te_W1, te_b1, te_W2, te_b2, proj_Wu, proj_bu, hu0, NNODE);
  k_node_init<32><<<nodeBlocks, 128, shm_init_i, stream>>>(
      x_item, t_item, te_W1, te_b1, te_W2, te_b2, proj_Wi, proj_bi, hi0, NNODE);

  // Stage 2: two SAGE layers, ping-pong.
  for (int l = 0; l < 2; ++l) {
    const float* cu = (l == 0) ? hu0 : hu1;
    const float* ci = (l == 0) ? hi0 : hi1;
    float* nu = (l == 0) ? hu1 : hu0;
    float* ni = (l == 0) ? hi1 : hi0;

    // user -> item
    k_zero<<<1024, 256, 0, stream>>>(mb, NF);
    k_zero<<<64, 256, 0, stream>>>(cntb, NNODE);
    k_scatter<<<scatBlocks, 256, 0, stream>>>(cu, e_ui_src, e_ui_dst, mb, cntb, NEDGE);
    k_sage<<<nodeBlocks, 128, shm_sage, stream>>>(
        mb, cntb, ci,
        sage_Wnbr + (l * 2 + 0) * 128 * 128,
        sage_Wroot + (l * 2 + 0) * 128 * 128,
        sage_b + (l * 2 + 0) * 128, ni, NNODE);

    // item -> user
    k_zero<<<1024, 256, 0, stream>>>(mb, NF);
    k_zero<<<64, 256, 0, stream>>>(cntb, NNODE);
    k_scatter<<<scatBlocks, 256, 0, stream>>>(ci, e_iu_src, e_iu_dst, mb, cntb, NEDGE);
    k_sage<<<nodeBlocks, 128, shm_sage, stream>>>(
        mb, cntb, cu,
        sage_Wnbr + (l * 2 + 1) * 128 * 128,
        sage_Wroot + (l * 2 + 1) * 128 * 128,
        sage_b + (l * 2 + 1) * 128, nu, NNODE);
  }
  // Final features in hu0 / hi0.

  // Stage 3: MLP heads (user: 128->256->256->64, item: 128->256->256->32).
  k_gemm<4, true><<<dim3(headBlocks, 4), 128, shm_gemm4, stream>>>(
      hu0, mlp_u_W1, mlp_u_b1, hid1, NHEAD, 128, 256);
  k_gemm<4, true><<<dim3(headBlocks, 4), 128, shm_gemm4, stream>>>(
      hid1, mlp_u_W2, mlp_u_b2, hid2, NHEAD, 256, 256);
  k_gemm<4, false><<<dim3(headBlocks, 1), 128, shm_gemm4, stream>>>(
      hid2, mlp_u_W3, mlp_u_b3, out, NHEAD, 256, 64);

  k_gemm<4, true><<<dim3(headBlocks, 4), 128, shm_gemm4, stream>>>(
      hi0, mlp_i_W1, mlp_i_b1, hid1, NHEAD, 128, 256);
  k_gemm<4, true><<<dim3(headBlocks, 4), 128, shm_gemm4, stream>>>(
      hid1, mlp_i_W2, mlp_i_b2, hid2, NHEAD, 256, 256);
  k_gemm<2, false><<<dim3(headBlocks, 1), 128, shm_gemm2, stream>>>(
      hid2, mlp_i_W3, mlp_i_b3, out + (long long)NHEAD * 64, NHEAD, 256, 32);
}
